// MesaNet_13752485282223
// MI455X (gfx1250) — matrix operand order repaired and checked by reading, not compiled, not run on hardware
//
#include <hip/hip_runtime.h>
#include <hip/hip_bf16.h>

// ---------------------------------------------------------------------------
// MesaNet forward for MI455X (gfx1250, wave32)
//   B=1, T=256, HID=2048, H=16, D=128, KD=2048, CONV=4, CG=30
// Strategy:
//   * 4 big GEMMs via v_wmma_f32_16x16x32_bf16, 128x128 block tiles staged in
//     LDS by the Tensor Data Mover (TENSORcnt-tracked async DMA)
//   * recurrence: 1 head per workgroup, H_kk/H_kv resident in LDS (128 KB)
// ---------------------------------------------------------------------------

#define T_LEN 256
#define HID   2048
#define NH    16
#define HD    128
#define KD    2048
#define CGIT  30

#define BM 128
#define BN 128
#define BK 32

typedef __attribute__((ext_vector_type(16))) __bf16 v16bf;
typedef __attribute__((ext_vector_type(8)))  float  v8f;

#if defined(__has_builtin)
#  if __has_builtin(__builtin_amdgcn_tensor_load_to_lds) && \
      __has_builtin(__builtin_amdgcn_s_wait_tensorcnt)
#    define HAVE_TDM 1
#  endif
#endif
#ifndef HAVE_TDM
#  define HAVE_TDM 0
#endif

#if HAVE_TDM
typedef unsigned int v4u __attribute__((ext_vector_type(4)));
typedef int          v8i __attribute__((ext_vector_type(8)));
typedef int          v4i __attribute__((ext_vector_type(4)));

// Issue one TDM 2D tile load: f32 tile (tile_w x tile_h) from a row-major
// tensor (row stride = stride_elems) into LDS at lds_off_bytes.
// D# packing per CDNA5 ISA 8.3 (group0) / 8.4 (group1).
// This toolchain's builtin takes 6 args (g0, g1, g2, g3, extra, cpol).
__device__ __forceinline__ void tdm_load_tile_f32(const float* gtile,
                                                  unsigned lds_off_bytes,
                                                  unsigned tile_w, unsigned tile_h,
                                                  unsigned tensor_w, unsigned tensor_h,
                                                  unsigned long long stride_elems) {
  const unsigned long long ga = (unsigned long long)(size_t)gtile;
  v4u g0;
  g0[0] = 1u;                                            // count=1, user-mode
  g0[1] = lds_off_bytes;                                 // lds_addr [63:32]
  g0[2] = (unsigned)(ga & 0xFFFFFFFFu);                  // global_addr lo
  g0[3] = ((unsigned)(ga >> 32) & 0x01FFFFFFu) | (2u << 30);  // addr hi | type=2
  v8i g1;
  g1[0] = (int)(2u << 16);                               // data_size=4B; mask=0
  g1[1] = (int)((tensor_w & 0xFFFFu) << 16);             // tensor_dim0[15:0]
  g1[2] = (int)((tensor_w >> 16) | ((tensor_h & 0xFFFFu) << 16));
  g1[3] = (int)((tensor_h >> 16) | (tile_w << 16));      // tile_dim0
  g1[4] = (int)(tile_h & 0xFFFFu);                       // tile_dim1; tile_dim2=0
  g1[5] = (int)(unsigned)(stride_elems & 0xFFFFFFFFull); // dim0_stride lo
  g1[6] = (int)(unsigned)((stride_elems >> 32) & 0xFFFFull);
  g1[7] = 0;
  const v4i z4 = {0, 0, 0, 0};
  const v8i z8 = {0, 0, 0, 0, 0, 0, 0, 0};
  __builtin_amdgcn_tensor_load_to_lds(g0, g1, z4, z4, z8, 0);
}
#endif  // HAVE_TDM

// ---------------------------------------------------------------------------
// GEMM: C[M,N] = A[M,K] @ B[K,N]; f32 in/out, bf16 WMMA compute (f32 accum).
// 256 threads (8 wave32) per block; block tile 128x128; K stepped by 32.
// Wave w -> 32x64 region: rows 32*(w>>1), cols 64*(w&1); 2x4 accumulators.
// Fragment layouts (A/B share contiguous k order, not the ISA table order):
//   A (16x32 bf16): lane {r=lane&15 -> M, kh=lane>>4}; elem e -> K=kh*16+e
//   B (32x16 bf16): lane -> N=lane&15, kh=lane>>4; elem e -> K=kh*16+e
//   C/D (16x16 f32): lane -> N=lane&15; vgpr j -> M=(lane>>4)*8+j
// ---------------------------------------------------------------------------
__global__ void __launch_bounds__(256)
gemm_bf16_wmma(const float* __restrict__ A, const float* __restrict__ B,
               float* __restrict__ C, int M, int N, int K) {
  __shared__ float Asm[BM * BK];   // 16 KB, [m][k]
  __shared__ float Bsm[BK * BN];   // 16 KB, [k][n]

  const int tid  = threadIdx.x;
  const int wid  = tid >> 5;
  const int lane = tid & 31;
  const int r    = lane & 15;
  const int kh   = lane >> 4;

  const int mb   = M / BM;
  const int bm   = (blockIdx.x % mb) * BM;
  const int bn   = (blockIdx.x / mb) * BN;
  const int wm   = (wid >> 1) * 32;   // 0,32,64,96
  const int wn   = (wid & 1) * 64;    // 0,64

  v8f acc[2][4];
  const v8f vzero = {};
#pragma unroll
  for (int i = 0; i < 2; ++i)
#pragma unroll
    for (int j = 0; j < 4; ++j) acc[i][j] = vzero;

  for (int kb = 0; kb < K; kb += BK) {
    __syncthreads();   // previous tile fully consumed before overwrite
#if HAVE_TDM
    if (wid == 0) {
      tdm_load_tile_f32(A + (size_t)bm * K + kb, (unsigned)(size_t)(void*)Asm,
                        /*tile_w=*/BK, /*tile_h=*/BM,
                        /*tensor_w=*/(unsigned)K, /*tensor_h=*/(unsigned)M,
                        (unsigned long long)K);
      tdm_load_tile_f32(B + (size_t)kb * N + bn, (unsigned)(size_t)(void*)Bsm,
                        /*tile_w=*/BN, /*tile_h=*/BK,
                        /*tensor_w=*/(unsigned)N, /*tensor_h=*/(unsigned)K,
                        (unsigned long long)N);
      __builtin_amdgcn_s_wait_tensorcnt(0);
    }
#else
    for (int i = tid; i < BM * BK; i += 256) {
      const int m = i / BK, k = i % BK;
      Asm[i] = A[(size_t)(bm + m) * K + kb + k];
    }
    for (int i = tid; i < BK * BN; i += 256) {
      const int k = i / BN, n = i % BN;
      Bsm[i] = B[(size_t)(kb + k) * N + bn + n];
    }
#endif
    __syncthreads();

    v16bf afrag[2], bfrag[4];
#pragma unroll
    for (int s = 0; s < 2; ++s) {
      const float* Ar = Asm + (wm + s * 16 + r) * BK + kh * 16;
#pragma unroll
      for (int e = 0; e < 8; ++e) {
        afrag[s][e]     = (__bf16)Ar[e];        // K = kh*16 + e
        afrag[s][e + 8] = (__bf16)Ar[e + 8];    // K = kh*16 + 8 + e
      }
    }
#pragma unroll
    for (int s = 0; s < 4; ++s) {
      const float* Bc = Bsm + (kh * 16) * BN + wn + s * 16 + r;
#pragma unroll
      for (int e = 0; e < 16; ++e) bfrag[s][e] = (__bf16)Bc[e * BN];
    }
#pragma unroll
    for (int i = 0; i < 2; ++i)
#pragma unroll
      for (int j = 0; j < 4; ++j)
        acc[i][j] = __builtin_amdgcn_wmma_f32_16x16x32_bf16(
            false, afrag[i], false, bfrag[j], (short)0, acc[i][j], false, false);
  }

#pragma unroll
  for (int i = 0; i < 2; ++i)
#pragma unroll
    for (int j = 0; j < 4; ++j) {
      float* Cp = C + (size_t)(bm + wm + i * 16 + kh * 8) * N + bn + wn + j * 16 + r;
#pragma unroll
      for (int jj = 0; jj < 8; ++jj) Cp[(size_t)jj * N] = acc[i][j][jj];
    }
}

// ---------------------------------------------------------------------------
// Depthwise causal conv (k=4) + SiLU + per-head l2norm.
// grid = T*NH blocks of 128 threads (one per channel within the head).
// ---------------------------------------------------------------------------
__global__ void conv_silu_l2norm(const float* __restrict__ lin,
                                 const float* __restrict__ w,
                                 float* __restrict__ out) {
  const int t = blockIdx.x / NH;
  const int h = blockIdx.x % NH;
  const int d = threadIdx.x;           // 0..127
  const int c = h * HD + d;

  float acc = 0.f;
#pragma unroll
  for (int i = 0; i < 4; ++i) {
    int ts = t - 3 + i;
    if (ts >= 0) acc += lin[(size_t)ts * KD + c] * w[c * 4 + i];
  }
  float y = acc / (1.f + __expf(-acc));          // SiLU

  __shared__ float red[4];
  float ss = y * y;
#pragma unroll
  for (int off = 16; off > 0; off >>= 1) ss += __shfl_xor(ss, off, 32);
  if ((threadIdx.x & 31) == 0) red[threadIdx.x >> 5] = ss;
  __syncthreads();
  float tot = red[0] + red[1] + red[2] + red[3];
  out[(size_t)t * KD + c] = y * rsqrtf(tot + 1e-6f);
}

// ---------------------------------------------------------------------------
// beta = sigmoid(x@Wb + bb), g = log_sigmoid(x@Wa + ba). One thread per (t,h).
// ---------------------------------------------------------------------------
__global__ void gates_kernel(const float* __restrict__ x,
                             const float* __restrict__ Wa, const float* __restrict__ ba,
                             const float* __restrict__ Wb, const float* __restrict__ bb,
                             float* __restrict__ g, float* __restrict__ beta) {
  const int idx = blockIdx.x * blockDim.x + threadIdx.x;
  if (idx >= T_LEN * NH) return;
  const int t = idx / NH, h = idx % NH;
  const float* xr = x + (size_t)t * HID;
  float sa = ba[h], sb = bb[h];
  for (int kk = 0; kk < HID; ++kk) {
    float xv = xr[kk];
    sa += xv * Wa[kk * NH + h];
    sb += xv * Wb[kk * NH + h];
  }
  g[idx]    = fminf(sa, 0.f) - log1pf(__expf(-fabsf(sa)));
  beta[idx] = 1.f / (1.f + __expf(-sb));
}

// ---------------------------------------------------------------------------
// Block-wide reduction of 128 per-thread values (tid<128 contribute).
// Result -> scal[dst]. All 256 threads must call.
// ---------------------------------------------------------------------------
__device__ __forceinline__ void blockReduce128(float val, float* scal, int tid, int dst) {
#pragma unroll
  for (int off = 16; off > 0; off >>= 1) val += __shfl_xor(val, off, 32);
  __syncthreads();                         // protect prior readers of scal[]
  if (tid < 128 && (tid & 31) == 0) scal[2 + (tid >> 5)] = val;
  __syncthreads();
  if (tid == 0) scal[dst] = scal[2] + scal[3] + scal[4] + scal[5];
  __syncthreads();
}

// ---------------------------------------------------------------------------
// Mesa recurrence: 1 block per head, 256 threads (8 wave32), state in LDS.
// LDS: Hkk 64KB + Hkv 64KB + vectors ~5KB  (CDNA5 WGP has 320KB LDS)
// ---------------------------------------------------------------------------
#define MESA_LDS_FLOATS (16384 + 16384 + 128 * 8 + 256 + 16)

__global__ void mesa_recurrence_kernel(const float* __restrict__ q,
                                       const float* __restrict__ k,
                                       const float* __restrict__ v,
                                       const float* __restrict__ g,
                                       const float* __restrict__ beta,
                                       const float* __restrict__ lambda_params,
                                       const float* __restrict__ o_norm_w,
                                       float* __restrict__ o) {
  extern __shared__ float lds[];
  float* Hkk  = lds;             // 16384
  float* Hkv  = Hkk + 16384;     // 16384
  float* lamb = Hkv + 16384;     // 128
  float* qv   = lamb + 128;
  float* kv   = qv + 128;
  float* vv   = kv + 128;
  float* xv   = vv + 128;        // CG solution
  float* rv   = xv + 128;
  float* pv   = rv + 128;
  float* Ap   = pv + 128;
  float* part = Ap + 128;        // 256 matvec partials
  float* scal = part + 256;      // [0]=rs [1]=tmp [2..5]=wave partials

  const int h = blockIdx.x;
  const int tid = threadIdx.x;

  for (int i = tid; i < 16384; i += 256) { Hkk[i] = 0.f; Hkv[i] = 0.f; }
  if (tid < 128) {
    float lp = lambda_params[h * HD + tid];
    float sp = (lp > 20.f) ? lp : log1pf(__expf(lp));   // softplus
    lamb[tid] = sp + 0.25f;                              // + lower bound
  }
  __syncthreads();

  for (int t = 0; t < T_LEN; ++t) {
    const size_t base = (size_t)t * KD + h * HD;
    if (tid < 128) {
      qv[tid] = q[base + tid];
      kv[tid] = k[base + tid];
      vv[tid] = v[base + tid];
    }
    __syncthreads();

    const float decay = __expf(g[t * NH + h]);
    const float bt    = beta[t * NH + h];

    for (int e = tid; e < 16384; e += 256) {
      const int i = e >> 7, j = e & 127;
      const float bki = bt * kv[i];
      Hkk[e] = Hkk[e] * decay + bki * kv[j];
      Hkv[e] = Hkv[e] * decay + bki * vv[j];
    }
    __syncthreads();

    // --- CG: (Hkk + diag(lamb)) x = qv, x0 = 0 ---
    float pr = 0.f;
    if (tid < 128) {
      const float b0 = qv[tid];
      xv[tid] = 0.f; rv[tid] = b0; pv[tid] = b0;
      pr = b0 * b0;
    }
    blockReduce128(pr, scal, tid, 0);                    // scal[0] = rs

    const int iRow  = tid & 127;
    const int half  = tid >> 7;
    for (int it = 0; it < CGIT; ++it) {
      const float* row = Hkk + iRow * 128 + half * 64;
      const float* ph  = pv + half * 64;
      float s = 0.f;
#pragma unroll 8
      for (int jj = 0; jj < 64; ++jj) {
        const int j = (jj + iRow) & 63;                  // spread LDS banks
        s += row[j] * ph[j];
      }
      part[tid] = s;
      __syncthreads();

      float pap = 0.f;
      if (tid < 128) {
        const float apv = part[tid] + part[tid + 128] + lamb[tid] * pv[tid];
        Ap[tid] = apv;
        pap = pv[tid] * apv;
      }
      blockReduce128(pap, scal, tid, 1);                 // scal[1] = pAp
      const float alpha = scal[0] / (scal[1] + 1e-12f);

      float rsn = 0.f;
      if (tid < 128) {
        xv[tid] += alpha * pv[tid];
        const float rn = rv[tid] - alpha * Ap[tid];
        rv[tid] = rn;
        rsn = rn * rn;
      }
      blockReduce128(rsn, scal, tid, 1);                 // scal[1] = rs_new
      if (tid < 128) {
        const float bcg = scal[1] / (scal[0] + 1e-12f);
        pv[tid] = rv[tid] + bcg * pv[tid];
      }
      __syncthreads();
      if (tid == 0) scal[0] = scal[1];
      __syncthreads();
    }

    // o_t[j] = sum_i x[i] * Hkv[i][j]
    {
      const int j = tid & 127;
      float s = 0.f;
#pragma unroll 8
      for (int ii = 0; ii < 64; ++ii) {
        const int i = half * 64 + ii;
        s += xv[i] * Hkv[i * 128 + j];
      }
      part[tid] = s;
      __syncthreads();

      float ss = 0.f, ov = 0.f;
      if (tid < 128) {
        ov = part[tid] + part[tid + 128];
        ss = ov * ov;
      }
      blockReduce128(ss, scal, tid, 1);                  // sum(o^2)
      if (tid < 128) {
        const float inv = rsqrtf(scal[1] * (1.f / 128.f) + 1e-5f);
        o[base + tid] = ov * inv * o_norm_w[tid];
      }
      __syncthreads();
    }
  }
}

// ---------------------------------------------------------------------------
extern "C" void kernel_launch(void* const* d_in, const int* in_sizes, int n_in,
                              void* d_out, int out_size, void* d_ws, size_t ws_size,
                              hipStream_t stream) {
  (void)in_sizes; (void)n_in; (void)out_size; (void)ws_size;

  const float* x   = (const float*)d_in[0];
  const float* Wq  = (const float*)d_in[1];
  const float* Wk  = (const float*)d_in[2];
  const float* Wv  = (const float*)d_in[3];
  const float* Wa  = (const float*)d_in[4];
  const float* ba  = (const float*)d_in[5];
  const float* Wb  = (const float*)d_in[6];
  const float* bb  = (const float*)d_in[7];
  const float* cwq = (const float*)d_in[8];
  const float* cwk = (const float*)d_in[9];
  const float* lp  = (const float*)d_in[10];
  const float* onw = (const float*)d_in[11];
  const float* Wo  = (const float*)d_in[12];

  const size_t TK = (size_t)T_LEN * KD;   // 524288
  float* ws    = (float*)d_ws;
  float* q_lin = ws;
  float* k_lin = q_lin + TK;
  float* v_buf = k_lin + TK;
  float* q_buf = v_buf + TK;
  float* k_buf = q_buf + TK;
  float* o_buf = k_buf + TK;
  float* g_buf = o_buf + TK;              // T*NH = 4096
  float* b_buf = g_buf + (size_t)T_LEN * NH;

  // --- projections (bf16 WMMA, f32 accumulate, TDM-staged tiles) ---
  dim3 gblk(256);
  dim3 ggrid((T_LEN / BM) * (KD / BN));   // 2 x 16 = 32 blocks
  gemm_bf16_wmma<<<ggrid, gblk, 0, stream>>>(x, Wq, q_lin, T_LEN, KD, HID);
  gemm_bf16_wmma<<<ggrid, gblk, 0, stream>>>(x, Wk, k_lin, T_LEN, KD, HID);
  gemm_bf16_wmma<<<ggrid, gblk, 0, stream>>>(x, Wv, v_buf, T_LEN, KD, HID);

  // --- gates ---
  gates_kernel<<<(T_LEN * NH + 255) / 256, 256, 0, stream>>>(
      x, Wa, ba, Wb, bb, g_buf, b_buf);

  // --- conv + SiLU + l2norm on q, k ---
  conv_silu_l2norm<<<T_LEN * NH, 128, 0, stream>>>(q_lin, cwq, q_buf);
  conv_silu_l2norm<<<T_LEN * NH, 128, 0, stream>>>(k_lin, cwk, k_buf);

  // --- recurrence: one head per WGP, state in LDS ---
  const int smem = MESA_LDS_FLOATS * (int)sizeof(float);   // ~133 KB
  (void)hipFuncSetAttribute((const void*)mesa_recurrence_kernel,
                            hipFuncAttributeMaxDynamicSharedMemorySize, smem);
  mesa_recurrence_kernel<<<NH, 256, smem, stream>>>(
      q_buf, k_buf, v_buf, g_buf, b_buf, lp, onw, o_buf);

  // --- output projection ---
  gemm_bf16_wmma<<<ggrid, gblk, 0, stream>>>(o_buf, Wo, (float*)d_out,
                                             T_LEN, HID, KD);
}
